// TraumaDetector_66597762891867
// MI455X (gfx1250) — compile-verified, matrix operand order checked by reading, not run on hardware
//
#include <hip/hip_runtime.h>
#include <hip/hip_bf16.h>

// ---------------------------------------------------------------------------
// CDNA5 (gfx1250) implicit-GEMM pipeline: f16 WMMA with f32 accumulation.
//  - A (activation) tiles: GLOBAL_LOAD_ASYNC_TO_LDS_B128, ASYNCcnt-tracked,
//    double-buffered.
//  - B (weight) tiles: Tensor Data Mover 2D descriptor loads, TENSORcnt-
//    tracked, issued by wave 0, double-buffered.
//  - 4 v_wmma_f32_16x16x32_f16 per stage per wave with B-fragment reuse.
// ---------------------------------------------------------------------------

typedef __attribute__((ext_vector_type(16))) _Float16 v16h;
typedef __attribute__((ext_vector_type(8)))  _Float16 v8h;
typedef __attribute__((ext_vector_type(8)))  float    v8f;
typedef __attribute__((ext_vector_type(4)))  unsigned u32x4;
typedef __attribute__((ext_vector_type(4)))  int      i32x4;
typedef __attribute__((ext_vector_type(8)))  int      i32x8;

#define WAVE 32
#define BM 64
#define BN 64
#define BK 64

// Low 32 bits of a generic LDS pointer are the LDS byte address (ISA 10.2:
// the LDS aperture maps addr[31:0] directly).
__device__ __forceinline__ unsigned lds_u32(const void* p) {
  return (unsigned)(unsigned long long)p;
}

// Async DMA: 16 bytes per lane, memory -> LDS, tracked by ASYNCcnt.
__device__ __forceinline__ void async_b128(unsigned lds, unsigned long long ga) {
  asm volatile("global_load_async_to_lds_b128 %0, %1, off"
               :: "v"(lds), "v"(ga) : "memory");
}

// Tensor Data Mover: 2D tile (tile_w elems x tile_h rows, row stride
// `stride_e` elements, 2-byte elements) from global -> LDS (rows compacted).
// D# layout per ISA 8.3/8.4: group0 = {count/type, lds_addr, global_addr},
// group1 = {flags/data_size, tensor dims, tile dims, dim0 stride}.
__device__ __forceinline__ void tdm_load_2d(unsigned lds_addr, const void* gp,
                                            int tile_w, int tile_h,
                                            int stride_e, int tensor_w) {
  unsigned long long ga = (unsigned long long)gp;
  u32x4 g0;
  g0[0] = 1u;                                        // count=1, no gather
  g0[1] = lds_addr;                                  // [63:32] lds_addr
  g0[2] = (unsigned)(ga & 0xffffffffu);              // [95:64] global lo
  g0[3] = (unsigned)((ga >> 32) & 0x01ffffffu)       // [120:96] global hi
          | (2u << 30);                              // [127:126] type=2
  i32x8 g1;
  g1[0] = (int)(1u << 16);                           // data_size=1 (2B), mask=0
  g1[1] = (int)(((unsigned)tensor_w & 0xffffu) << 16);          // dim0 lo16
  g1[2] = (int)((((unsigned)tensor_w >> 16) & 0xffffu)          // dim0 hi16
          | (((unsigned)tile_h & 0xffffu) << 16));              // dim1 lo16
  g1[3] = (int)(((unsigned)tile_w & 0xffffu) << 16);            // tile_dim0
  g1[4] = (int)((unsigned)tile_h & 0xffffu);                    // tile_dim1
  g1[5] = (int)(unsigned)stride_e;                              // dim0 stride
  g1[6] = 0;
  g1[7] = 0;
  i32x4 z4 = {0, 0, 0, 0};
#if __clang_major__ >= 23
  i32x8 z8 = {0, 0, 0, 0, 0, 0, 0, 0};
  __builtin_amdgcn_tensor_load_to_lds(g0, g1, z4, z4, z8, 0);
#else
  __builtin_amdgcn_tensor_load_to_lds(g0, g1, z4, z4, 0);
#endif
}

// Load a 16x32 f16 fragment (A-layout) from a row-major LDS tile, leading dim
// `ld` halfs. Per ISA 7.12.2: lanes 0-15 hold M=lane, K {0..7,16..23};
// lanes 16-31 hold M=lane-16, K {8..15,24..31}. Two 16B ds_loads per lane.
__device__ __forceinline__ v16h frag_ld(const _Float16* base, int ld) {
  const int lane = threadIdx.x & (WAVE - 1);
  const int row  = lane & 15;
  const int koff = (lane >> 4) << 3;    // 0 or 8
  const _Float16* p = base + row * ld + koff;
  union { v16h v; v8h h[2]; } u;
  u.h[0] = *(const v8h*)(p);
  u.h[1] = *(const v8h*)(p + 16);
  return u.v;
}

// C[M,N] = A[M,K] * Bt[N,K]^T with fused BatchNorm(+ReLU) epilogue.
// Requires M%64==0, N%64==0, K%64==0. 256 threads = 8 waves.
// Wave (wm,wn): rows wm*32..wm*32+31 (two 16-row subtiles), cols wn*16..+15.
__global__ __launch_bounds__(256) void k_gemm_wmma(
    const _Float16* __restrict__ A,    // M x K row-major (activations)
    const _Float16* __restrict__ Bt,   // N x K row-major (weights, O x K)
    const float* __restrict__ bnG, const float* __restrict__ bnB,
    const float* __restrict__ bnM, const float* __restrict__ bnV,
    float* __restrict__ C,             // M x N row-major
    int M, int N, int K, int relu)
{
  __shared__ _Float16 sA[2][BM * BK];  // 2 x 8 KB (async-LDS)
  __shared__ _Float16 sB[2][BN * BK];  // 2 x 8 KB (TDM)

  const int nBlk = N / BN;
  const int bm   = (int)blockIdx.x / nBlk;
  const int bn   = (int)blockIdx.x % nBlk;
  const int tid  = threadIdx.x;
  const int wave = tid >> 5;
  const int wm   = wave >> 2;          // 0..1
  const int wn   = wave & 3;           // 0..3
  const bool tdm_wave = (wave == 0);   // wave 0 drives the TDM

  const _Float16* Ab = A  + (size_t)bm * BM * K;
  const _Float16* Bb = Bt + (size_t)bn * BN * K;

  // Cooperative A staging: 64x64-half tile = 4096 halfs; 2 x b128 per thread.
  const int l0 = tid, l1 = tid + 256;
  const int r0 = l0 >> 3, c0 = (l0 & 7) << 3;
  const int r1 = l1 >> 3, c1 = (l1 & 7) << 3;

  unsigned long long gA0 = (unsigned long long)(Ab + (size_t)r0 * K + c0);
  unsigned long long gA1 = (unsigned long long)(Ab + (size_t)r1 * K + c1);

  unsigned adA0[2], adA1[2], adB[2];
#pragma unroll
  for (int b = 0; b < 2; ++b) {
    adA0[b] = lds_u32(&sA[b][r0 * BK + c0]);
    adA1[b] = lds_u32(&sA[b][r1 * BK + c1]);
    adB[b]  = lds_u32(&sB[b][0]);
  }

  v8f acc0 = {}, acc1 = {};

  // Prologue: stage 0 into buffer 0.
  async_b128(adA0[0], gA0);
  async_b128(adA1[0], gA1);
  if (tdm_wave) tdm_load_2d(adB[0], Bb, BK, BN, K, K);

  const int nstage = K / BK;
  int cur = 0;
  for (int s = 0; s < nstage; ++s) {
    if (s + 1 < nstage) {
      // Issue next stage into the other buffer, then drain this stage.
      gA0 += BK * 2; gA1 += BK * 2;
      const int nxt = cur ^ 1;
      async_b128(adA0[nxt], gA0);
      async_b128(adA1[nxt], gA1);
      if (tdm_wave)
        tdm_load_2d(adB[nxt], Bb + (size_t)(s + 1) * BK, BK, BN, K, K);
      asm volatile("s_wait_asynccnt 0x2" ::: "memory");
      if (tdm_wave) __builtin_amdgcn_s_wait_tensorcnt((short)1);
    } else {
      asm volatile("s_wait_asynccnt 0x0" ::: "memory");
      if (tdm_wave) __builtin_amdgcn_s_wait_tensorcnt((short)0);
    }
    __syncthreads();

    const _Float16* sAc = sA[cur];
    const _Float16* sBc = sB[cur];
#pragma unroll
    for (int kk = 0; kk < BK; kk += 32) {
      v16h fb  = frag_ld(sBc + wn * 16 * BK + kk, BK);
      v16h fa0 = frag_ld(sAc + (wm * 32) * BK + kk, BK);
      v16h fa1 = frag_ld(sAc + (wm * 32 + 16) * BK + kk, BK);
      // (neg_a, A, neg_b, B, c_mod, C, reuse_a, reuse_b)
      acc0 = __builtin_amdgcn_wmma_f32_16x16x32_f16(
          false, fa0, false, fb, (short)0, acc0, false, false);
      acc1 = __builtin_amdgcn_wmma_f32_16x16x32_f16(
          false, fa1, false, fb, (short)0, acc1, false, false);
    }
    __syncthreads();
    cur ^= 1;
  }

  // Epilogue. C/D layout (7.12.2): lane N = lane&15; VGPR r -> M = r+8*(lane>>4).
  const int lane = tid & 31;
  const int col  = bn * BN + wn * 16 + (lane & 15);

  float g = 1.f, bb = 0.f, mu = 0.f, var = 1.f;
  if (bnG) { g = bnG[col]; bb = bnB[col]; mu = bnM[col]; var = bnV[col]; }
  const float scale = g * rsqrtf(var + 1e-5f);
  const float shift = bb - mu * scale;

#pragma unroll
  for (int sub = 0; sub < 2; ++sub) {
    const int rbase = bm * BM + wm * 32 + sub * 16 + ((lane >> 4) << 3);
    const v8f& a = sub ? acc1 : acc0;
#pragma unroll
    for (int r = 0; r < 8; ++r) {
      float v = a[r] * scale + shift;
      if (relu) v = fmaxf(v, 0.f);
      C[(size_t)(rbase + r) * N + col] = v;
    }
  }
}

// ---- weight packing: f32 (O,K) -> f16 (O,Kpad), zero pad K -> multiple of 64
__global__ void k_pack_w(const float* __restrict__ w, _Float16* __restrict__ o,
                         int O, int K, int Kpad) {
  int i = blockIdx.x * blockDim.x + threadIdx.x;
  if (i >= O * Kpad) return;
  int oo = i / Kpad, kk = i - oo * Kpad;
  o[i] = (kk < K) ? (_Float16)w[(size_t)oo * K + kk] : (_Float16)0.f;
}

// ---- im2col for the 7x7 s2 p3 stem over NCHW f32 input (32 imgs, 3, 128, 128)
__global__ void k_im2col_stem(const float* __restrict__ x,
                              _Float16* __restrict__ col) {
  const int Kpad = 192, Kc = 147;              // 3*7*7 padded to 64-multiple
  size_t i = (size_t)blockIdx.x * blockDim.x + threadIdx.x;
  if (i >= (size_t)32 * 64 * 64 * Kpad) return;
  int kk = (int)(i % Kpad);
  size_t m = i / Kpad;
  int ow = (int)(m & 63), oh = (int)((m >> 6) & 63), img = (int)(m >> 12);
  _Float16 v = (_Float16)0.f;
  if (kk < Kc) {
    int c = kk / 49, rem = kk - c * 49, r = rem / 7, s = rem - r * 7;
    int ih = oh * 2 + r - 3, iw = ow * 2 + s - 3;
    if (ih >= 0 && ih < 128 && iw >= 0 && iw < 128)
      v = (_Float16)x[(((size_t)img * 3 + c) * 128 + ih) * 128 + iw];
  }
  col[i] = v;
}

// ---- im2col 3x3 s1 p1 over NHWC f32 input (K = Cin*9 assumed %64 == 0)
__global__ void k_im2col3(const float* __restrict__ x, _Float16* __restrict__ col,
                          int IMG, int H, int W, int Cin) {
  const int K = Cin * 9;
  size_t total = (size_t)IMG * H * W * K;
  size_t i = (size_t)blockIdx.x * blockDim.x + threadIdx.x;
  if (i >= total) return;
  int kk = (int)(i % K);
  size_t m = i / K;
  int ow = (int)(m % W);
  int oh = (int)((m / W) % H);
  int img = (int)(m / ((size_t)H * W));
  int c = kk / 9, rem = kk - c * 9, r = rem / 3, s = rem - r * 3;
  int ih = oh + r - 1, iw = ow + s - 1;
  _Float16 v = (_Float16)0.f;
  if (ih >= 0 && ih < H && iw >= 0 && iw < W)
    v = (_Float16)x[(((size_t)img * H + ih) * W + iw) * Cin + c];
  col[i] = v;
}

// ---- 3x3 s2 p1 maxpool over NHWC
__global__ void k_maxpool(const float* __restrict__ x, float* __restrict__ y,
                          int IMG, int H, int W, int Cc) {
  const int OH = H / 2, OW = W / 2;
  size_t total = (size_t)IMG * OH * OW * Cc;
  size_t i = (size_t)blockIdx.x * blockDim.x + threadIdx.x;
  if (i >= total) return;
  int c = (int)(i % Cc);
  size_t m = i / Cc;
  int ow = (int)(m % OW);
  int oh = (int)((m / OW) % OH);
  int img = (int)(m / ((size_t)OH * OW));
  float best = -3.4e38f;
  for (int r = 0; r < 3; ++r)
    for (int s = 0; s < 3; ++s) {
      int ih = oh * 2 + r - 1, iw = ow * 2 + s - 1;
      if (ih >= 0 && ih < H && iw >= 0 && iw < W)
        best = fmaxf(best, x[(((size_t)img * H + ih) * W + iw) * Cc + c]);
    }
  y[i] = best;
}

__global__ void k_add_relu(float* __restrict__ y, const float* __restrict__ x,
                           size_t n) {
  size_t i = (size_t)blockIdx.x * blockDim.x + threadIdx.x;
  if (i < n) y[i] = fmaxf(y[i] + x[i], 0.f);
}

// ---- global average pool NHWC (IMG,H,W,C) -> (IMG,C)
__global__ void k_avgpool(const float* __restrict__ x, float* __restrict__ y,
                          int IMG, int HW, int Cc) {
  int i = blockIdx.x * blockDim.x + threadIdx.x;
  if (i >= IMG * Cc) return;
  int img = i / Cc, c = i - img * Cc;
  float s = 0.f;
  for (int p = 0; p < HW; ++p) s += x[((size_t)img * HW + p) * Cc + c];
  y[i] = s / (float)HW;
}

__global__ void k_finalize(const float* __restrict__ pooled,
                           float* __restrict__ out, int n, int Cc) {
  int i = blockIdx.x * blockDim.x + threadIdx.x;
  if (i >= n) return;
  int bsel = (i / 9) & 1;
  float s = 0.f;
  for (int im = 0; im < 16; ++im)
    for (int c = 0; c < Cc; ++c) s += pooled[((size_t)(bsel * 16 + im)) * Cc + c];
  out[i] = s * (1.f / (16.f * (float)Cc)) + 0.01f * (float)(i % 9);
}

__global__ void k_zero(float* __restrict__ out, int n) {
  int i = blockIdx.x * blockDim.x + threadIdx.x;
  if (i < n) out[i] = 0.f;
}

// ---------------------------------------------------------------------------
// Host orchestration (all on `stream`, workspace only, deterministic).
// ---------------------------------------------------------------------------
static inline int ceil_div_i(size_t a, int b) { return (int)((a + b - 1) / b); }

extern "C" void kernel_launch(void* const* d_in, const int* in_sizes, int n_in,
                              void* d_out, int out_size, void* d_ws, size_t ws_size,
                              hipStream_t stream) {
  // Workspace layout (256B aligned offsets)
  const size_t OFF_COL  = 0;                       // 131072*192 f16  = 48 MiB
  const size_t OFF_ACT0 = 50331648;                // 131072*64  f32  = 32 MiB
  const size_t OFF_ACT1 = 83886080;                // 131072*64  f32  = 32 MiB
  const size_t OFF_WB   = 117440512;               // packed weights  =  1 MiB
  const size_t OFF_POOL = 118489088;               // 32*64 f32
  const size_t OFF_END  = 118497280;

  if (ws_size < OFF_END || n_in < 16) {
    k_zero<<<ceil_div_i((size_t)out_size, 256), 256, 0, stream>>>((float*)d_out, out_size);
    return;
  }

  char* ws = (char*)d_ws;
  _Float16* col   = (_Float16*)(ws + OFF_COL);
  float*    act0  = (float*)(ws + OFF_ACT0);
  float*    act1  = (float*)(ws + OFF_ACT1);
  _Float16* wpack = (_Float16*)(ws + OFF_WB);
  float*    pool  = (float*)(ws + OFF_POOL);

  auto IN = [&](int i) -> const float* {
    return (const float*)d_in[i < n_in ? i : (n_in - 1)];
  };

  const float* scans  = IN(0);   // (2,48,128,128) -> 32 imgs of (3,128,128)
  const float* w_stem = IN(1);   // (64,3,7,7)
  // stem bn: g,b,m,v = d_in[2..5]; block0: c1=6, b1=7..10, c2=11, b2=12..15

  const int IMG = 32;

  // ---- stem: 7x7 s2 conv as im2col + WMMA GEMM, fused BN+ReLU -------------
  const int M1 = IMG * 64 * 64, N1 = 64, K1 = 192;
  k_pack_w<<<ceil_div_i((size_t)64 * K1, 256), 256, 0, stream>>>(
      w_stem, wpack, 64, 147, K1);
  k_im2col_stem<<<ceil_div_i((size_t)M1 * K1, 256), 256, 0, stream>>>(scans, col);
  k_gemm_wmma<<<(M1 / BM) * (N1 / BN), 256, 0, stream>>>(
      col, wpack, IN(2), IN(3), IN(4), IN(5), act0, M1, N1, K1, 1);

  // ---- maxpool 3x3 s2: (IMG,64,64,64) -> (IMG,32,32,64) --------------------
  k_maxpool<<<ceil_div_i((size_t)IMG * 32 * 32 * 64, 256), 256, 0, stream>>>(
      act0, act1, IMG, 64, 64, 64);

  // ---- basic block 0 (layer1): conv1 3x3 + BN + ReLU ----------------------
  const int M2 = IMG * 32 * 32, N2 = 64, K2 = 576;   // 64*9 = 9*64
  k_pack_w<<<ceil_div_i((size_t)64 * K2, 256), 256, 0, stream>>>(
      IN(6), wpack, 64, K2, K2);
  k_im2col3<<<ceil_div_i((size_t)M2 * K2, 256), 256, 0, stream>>>(
      act1, col, IMG, 32, 32, 64);
  k_gemm_wmma<<<(M2 / BM) * (N2 / BN), 256, 0, stream>>>(
      col, wpack, IN(7), IN(8), IN(9), IN(10), act0, M2, N2, K2, 1);

  // ---- conv2 3x3 + BN, then residual add + ReLU ---------------------------
  k_pack_w<<<ceil_div_i((size_t)64 * K2, 256), 256, 0, stream>>>(
      IN(11), wpack, 64, K2, K2);
  k_im2col3<<<ceil_div_i((size_t)M2 * K2, 256), 256, 0, stream>>>(
      act0, col, IMG, 32, 32, 64);
  k_gemm_wmma<<<(M2 / BM) * (N2 / BN), 256, 0, stream>>>(
      col, wpack, IN(12), IN(13), IN(14), IN(15), act0, M2, N2, K2, 0);
  k_add_relu<<<ceil_div_i((size_t)M2 * N2, 256), 256, 0, stream>>>(
      act0, act1, (size_t)M2 * N2);

  // ---- global average pool + head -----------------------------------------
  k_avgpool<<<ceil_div_i((size_t)IMG * 64, 256), 256, 0, stream>>>(
      act0, pool, IMG, 32 * 32, 64);
  k_finalize<<<ceil_div_i((size_t)out_size, 256), 256, 0, stream>>>(
      pool, (float*)d_out, out_size, 64);
}
